// CrossAttentionS2T_19542101197594
// MI455X (gfx1250) — matrix operand order repaired and checked by reading, not compiled, not run on hardware
//
#include <hip/hip_runtime.h>
#include <hip/hip_bf16.h>
#include <math.h>

// ---------------------------------------------------------------------------
// CDNA5 (gfx1250) bf16-WMMA space->time cross attention.
//   s = rearrange(s_x) + pos            (pack_s kernel, fp32 -> bf16)
//   q = (t_x @ q_w^T + q_b) * scale     (WMMA GEMM, bf16 out)
//   kv = s @ kv_w^T + kv_b              (WMMA GEMM, bf16 out)
//   o  = softmax(q k^T) v               (flash attention, transposed scores:
//                                        S^T = K Q^T so softmax is in-lane)
//   out = o @ proj_w^T + proj_b         (WMMA GEMM, fp32 out)
// ---------------------------------------------------------------------------

typedef __attribute__((ext_vector_type(16))) __bf16 v16bf;
typedef __attribute__((ext_vector_type(8)))  __bf16 v8bf;
typedef __attribute__((ext_vector_type(2)))  __bf16 v2bf;
typedef __attribute__((ext_vector_type(8)))  float  v8f;

#define DIM   768
#define HEADS 12
#define HD    64
#define BB    8
#define NQ    1576
#define NK    1576
#define NSP   197
#define TTF   8
#define QTILES 99  // ceil(1576/16)

__device__ __forceinline__ v8f v8f_zero() {
    v8f z;
#pragma unroll
    for (int i = 0; i < 8; ++i) z[i] = 0.0f;
    return z;
}

__device__ __forceinline__ v16bf cat16(v8bf a, v8bf b) {
    v16bf r;
#pragma unroll
    for (int i = 0; i < 8; ++i) { r[i] = a[i]; r[i + 8] = b[i]; }
    return r;
}

// ---- Fragment loaders (wave32; contiguous k order, same in A and B) --------------
// A 16x32 bf16: lane&15 = row M; half=lane>>4; elem e -> K=e+16*half
// => one contiguous 32B run at k0+16*half.
__device__ __forceinline__ v16bf load_a_frag(const __bf16* A, int ld, int row,
                                             int k0, int lane) {
    const int half = (lane >> 4) & 1;
    const __bf16* p = A + (size_t)row * ld + k0 + 16 * half;
    return cat16(*(const v8bf*)p, *(const v8bf*)(p + 8));
}

// B 32x16 bf16 with per-lane row pointer: lane&15 = col N; elem e -> K=e+16*half
// => one contiguous 32B run at rowptr + 16*half.
__device__ __forceinline__ v16bf load_b_frag_row(const __bf16* rowp, int lane) {
    const int half = (lane >> 4) & 1;
    const __bf16* p = rowp + 16 * half;
    return cat16(*(const v8bf*)p, *(const v8bf*)(p + 8));
}

// ---- Prep kernels ---------------------------------------------------------
__global__ void cast_f32_to_bf16(const float* __restrict__ in,
                                 __bf16* __restrict__ out, int n) {
    int i = blockIdx.x * blockDim.x + threadIdx.x;
    if (i < n) out[i] = (__bf16)in[i];
}

// s[b, tt*197+nn, d] = s_x[nn, b*8+tt, d] + pos[tt*197+nn, d]   -> bf16
__global__ void pack_s_kernel(const float* __restrict__ s_x,
                              const float* __restrict__ pos,
                              __bf16* __restrict__ out) {
    int i = blockIdx.x * blockDim.x + threadIdx.x;
    if (i >= BB * NK * DIM) return;
    int d = i % DIM;
    int rest = i / DIM;
    int kk = rest % NK;
    int b = rest / NK;
    int tt = kk / NSP;
    int nn = kk % NSP;
    float v = s_x[((size_t)(nn * (BB * TTF) + b * TTF + tt)) * DIM + d] +
              pos[(size_t)kk * DIM + d];
    out[i] = (__bf16)v;
}

// ---- Generic WMMA GEMM: out[M,N] = (A[M,K] @ W[N,K]^T + bias) * scale -----
// block = 256 threads = 8 waves arranged 2(M) x 4(N); wave tile 32x64
// (2 A-frags x 4 B-frags -> 8 WMMAs per 32-wide K step).
template <typename OutT>
__global__ void gemm_bf16_wmma(const __bf16* __restrict__ A,
                               const __bf16* __restrict__ W,
                               const float* __restrict__ bias,
                               OutT* __restrict__ out,
                               int M, int N, int K, float scale) {
    const int lane = threadIdx.x & 31;
    const int wave = threadIdx.x >> 5;
    const int wm = wave & 1, wn = wave >> 1;
    const int m0 = (blockIdx.x * 2 + wm) * 32;
    const int n0 = (blockIdx.y * 4 + wn) * 64;
    const int nn = lane & 15;
    const int rA0 = m0 + nn;
    const int rA1 = m0 + 16 + nn;

    v8f acc[2][4];
#pragma unroll
    for (int u = 0; u < 2; ++u)
#pragma unroll
        for (int t = 0; t < 4; ++t) acc[u][t] = v8f_zero();

    for (int k0 = 0; k0 < K; k0 += 32) {
        if (k0 + 32 < K) {  // prefetch next A tile (global_prefetch_b8)
            __builtin_prefetch(A + (size_t)rA0 * K + k0 + 32, 0, 3);
            __builtin_prefetch(A + (size_t)rA1 * K + k0 + 32, 0, 3);
        }
        const v16bf a0 = load_a_frag(A, K, rA0, k0, lane);
        const v16bf a1 = load_a_frag(A, K, rA1, k0, lane);
#pragma unroll
        for (int t = 0; t < 4; ++t) {
            const v16bf b =
                load_b_frag_row(W + (size_t)(n0 + t * 16 + nn) * K + k0, lane);
            acc[0][t] = __builtin_amdgcn_wmma_f32_16x16x32_bf16(
                false, a0, false, b, (short)0, acc[0][t], false, false);
            acc[1][t] = __builtin_amdgcn_wmma_f32_16x16x32_bf16(
                false, a1, false, b, (short)0, acc[1][t], false, false);
        }
    }

    const int half = (lane >> 4) & 1;
#pragma unroll
    for (int t = 0; t < 4; ++t) {
        const int col = n0 + t * 16 + nn;
        const float bv = bias[col];
#pragma unroll
        for (int u = 0; u < 2; ++u)
#pragma unroll
            for (int r = 0; r < 8; ++r) {
                const int mrow = m0 + u * 16 + r + 8 * half;
                out[(size_t)mrow * N + col] = (OutT)((acc[u][t][r] + bv) * scale);
            }
    }
}

// ---- Flash attention, transposed scores: one wave per (b, h, 16-q tile) ---
// S^T = K(16k x 64) @ Q^T(64 x 16q): lane&15 = query column, so the softmax
// key-reduction is in-lane over C elements + ONE shfl_xor(16) across halves.
// O^T = V^T(64 x 32k) @ P^T(32k x 16q); V^T staged in LDS (transposed write),
// P^T staged through 1KB LDS to reach the B-fragment layout.
__global__ void attn_wmma(const __bf16* __restrict__ qm,
                          const __bf16* __restrict__ kvm,
                          __bf16* __restrict__ om) {
    __shared__ __align__(16) __bf16 vt_s[8][HD * 32];  // per wave: V^T (hd,key)
    __shared__ __align__(16) __bf16 p_s[8][16 * 32];   // per wave: P (query,key)

    const int lane = threadIdx.x & 31;
    const int wave = threadIdx.x >> 5;
    const int gw = blockIdx.x * 8 + wave;
    const int qt = gw % QTILES;
    const int bh = gw / QTILES;
    const int h = bh % HEADS;
    const int b = bh / HEADS;

    const int half = (lane >> 4) & 1;
    const int nn = lane & 15;
    const int q0 = qt * 16;

    const __bf16* Qb = qm + (size_t)b * NQ * DIM + h * HD;
    const __bf16* Kb = kvm + (size_t)b * NK * (2 * DIM) + h * HD;
    const __bf16* Vb = Kb + DIM;
    __bf16* vt = vt_s[wave];
    __bf16* pp = p_s[wave];

    // Q as B operand (hd x query): load once; clamp padded query rows.
    int qrow = q0 + nn;
    if (qrow >= NQ) qrow = NQ - 1;
    const v16bf qb0 = load_b_frag_row(Qb + (size_t)qrow * DIM, lane);
    const v16bf qb1 = load_b_frag_row(Qb + (size_t)qrow * DIM + 32, lane);

    float m = -1e30f, l = 0.0f;
    v8f o[4];
#pragma unroll
    for (int t = 0; t < 4; ++t) o[t] = v8f_zero();

    const int vkp = lane & 15;  // key-pair index for V staging
    const int vs = lane >> 4;   // hd half for V staging

    for (int j = 0; j < NK; j += 32) {
        if (j + 32 < NK) {  // prefetch next chunk's K/V rows
            __builtin_prefetch(Kb + (size_t)(j + 32 + nn) * (2 * DIM), 0, 3);
            __builtin_prefetch(Kb + (size_t)(j + 48 + nn) * (2 * DIM), 0, 3);
            __builtin_prefetch(Vb + (size_t)(j + 32 + 2 * vkp) * (2 * DIM) + vs * 32,
                               0, 3);
        }

        // ---- stage V^T tile (hd x 32 keys) into LDS, coalesced b128 loads
        {
            int k0r = j + 2 * vkp;     if (k0r > NK - 1) k0r = NK - 1;
            int k1r = j + 2 * vkp + 1; if (k1r > NK - 1) k1r = NK - 1;
            union { uint4 q[4]; __bf16 e[32]; } U0, U1;
            const uint4* s0 = (const uint4*)(Vb + (size_t)k0r * (2 * DIM) + vs * 32);
            const uint4* s1 = (const uint4*)(Vb + (size_t)k1r * (2 * DIM) + vs * 32);
#pragma unroll
            for (int i = 0; i < 4; ++i) { U0.q[i] = s0[i]; U1.q[i] = s1[i]; }
#pragma unroll
            for (int hh = 0; hh < 32; ++hh) {
                v2bf pr;
                pr[0] = U0.e[hh];
                pr[1] = U1.e[hh];
                *(v2bf*)(vt + (vs * 32 + hh) * 32 + 2 * vkp) = pr;
            }
        }
        __builtin_amdgcn_wave_barrier();

        // ---- S^T tiles: c0 = keys j..j+15, c1 = keys j+16..j+31
        v8f c0 = v8f_zero(), c1 = v8f_zero();
        {
            int kr0 = j + nn;      if (kr0 > NK - 1) kr0 = NK - 1;
            int kr1 = j + 16 + nn; if (kr1 > NK - 1) kr1 = NK - 1;
            v16bf ka = load_a_frag(Kb, 2 * DIM, kr0, 0, lane);
            c0 = __builtin_amdgcn_wmma_f32_16x16x32_bf16(false, ka, false, qb0,
                                                         (short)0, c0, false, false);
            ka = load_a_frag(Kb, 2 * DIM, kr0, 32, lane);
            c0 = __builtin_amdgcn_wmma_f32_16x16x32_bf16(false, ka, false, qb1,
                                                         (short)0, c0, false, false);
            ka = load_a_frag(Kb, 2 * DIM, kr1, 0, lane);
            c1 = __builtin_amdgcn_wmma_f32_16x16x32_bf16(false, ka, false, qb0,
                                                         (short)0, c1, false, false);
            ka = load_a_frag(Kb, 2 * DIM, kr1, 32, lane);
            c1 = __builtin_amdgcn_wmma_f32_16x16x32_bf16(false, ka, false, qb1,
                                                         (short)0, c1, false, false);
        }

        // ---- mask tail keys, in-lane max over 16 elements
        float mx = -1e30f;
#pragma unroll
        for (int r = 0; r < 8; ++r) {
            const bool ok0 = (j + r + 8 * half) < NK;
            const bool ok1 = (j + 16 + r + 8 * half) < NK;
            c0[r] = ok0 ? c0[r] : -1e30f;
            c1[r] = ok1 ? c1[r] : -1e30f;
            mx = fmaxf(mx, fmaxf(c0[r], c1[r]));
        }
        mx = fmaxf(mx, __shfl_xor(mx, 16, 32));  // combine wave halves
        const float mnew = fmaxf(m, mx);
        const float alpha = __expf(m - mnew);

        float p0[8], p1[8];
        float rs = 0.0f;
#pragma unroll
        for (int r = 0; r < 8; ++r) {
            p0[r] = __expf(c0[r] - mnew);  // masked scores -> exp(-1e30) = 0
            p1[r] = __expf(c1[r] - mnew);
            rs += p0[r] + p1[r];
        }
        rs += __shfl_xor(rs, 16, 32);
        l = l * alpha + rs;
        m = mnew;
#pragma unroll
        for (int t = 0; t < 4; ++t)
#pragma unroll
            for (int r = 0; r < 8; ++r) o[t][r] *= alpha;

        // ---- stage P (query, key) to LDS as bf16, packed b32 stores
#pragma unroll
        for (int r = 0; r < 8; r += 2) {
            v2bf w0, w1;
            w0[0] = (__bf16)p0[r]; w0[1] = (__bf16)p0[r + 1];
            w1[0] = (__bf16)p1[r]; w1[1] = (__bf16)p1[r + 1];
            *(v2bf*)(pp + nn * 32 + 8 * half + r) = w0;
            *(v2bf*)(pp + nn * 32 + 16 + 8 * half + r) = w1;
        }
        __builtin_amdgcn_wave_barrier();

        // ---- O^T += V^T @ P^T
        const v16bf pbf = load_b_frag_row(pp + nn * 32, lane);
#pragma unroll
        for (int t = 0; t < 4; ++t) {
            v16bf va = load_a_frag(vt, 32, t * 16 + nn, 0, lane);
            o[t] = __builtin_amdgcn_wmma_f32_16x16x32_bf16(
                false, va, false, pbf, (short)0, o[t], false, false);
        }
        __builtin_amdgcn_wave_barrier();
    }

    // ---- store: O^T element (t,r) -> hd = t*16 + 8*half + r, query = nn
    if (q0 + nn < NQ) {
        const float inv = 1.0f / l;
        __bf16* orow = om + (size_t)(b * NQ + q0 + nn) * DIM + h * HD;
#pragma unroll
        for (int t = 0; t < 4; ++t) {
            v8bf w;
#pragma unroll
            for (int r = 0; r < 8; ++r) w[r] = (__bf16)(o[t][r] * inv);
            *(v8bf*)(orow + t * 16 + 8 * half) = w;  // contiguous 16B store
        }
    }
}

// ---------------------------------------------------------------------------
extern "C" void kernel_launch(void* const* d_in, const int* in_sizes, int n_in,
                              void* d_out, int out_size, void* d_ws, size_t ws_size,
                              hipStream_t stream) {
    (void)in_sizes; (void)n_in; (void)out_size; (void)ws_size;
    const float* s_x    = (const float*)d_in[0];
    const float* t_x    = (const float*)d_in[1];
    const float* pos    = (const float*)d_in[2];
    const float* q_w    = (const float*)d_in[3];
    const float* q_b    = (const float*)d_in[4];
    const float* kv_w   = (const float*)d_in[5];
    const float* kv_b   = (const float*)d_in[6];
    const float* proj_w = (const float*)d_in[7];
    const float* proj_b = (const float*)d_in[8];
    float* out = (float*)d_out;

    const size_t M = (size_t)BB * NQ;  // 12608

    size_t off = 0;
    auto carve = [&](size_t bytes) {
        void* p = (char*)d_ws + off;
        off += (bytes + 255) & ~(size_t)255;
        return p;
    };
    __bf16* s_bf   = (__bf16*)carve((size_t)BB * NK * DIM * 2);
    __bf16* tx_bf  = (__bf16*)carve(M * DIM * 2);
    __bf16* qw_bf  = (__bf16*)carve((size_t)DIM * DIM * 2);
    __bf16* kvw_bf = (__bf16*)carve((size_t)2 * DIM * DIM * 2);
    __bf16* pw_bf  = (__bf16*)carve((size_t)DIM * DIM * 2);
    __bf16* q_bf   = (__bf16*)carve(M * DIM * 2);
    __bf16* kv_bf  = (__bf16*)carve(M * 2 * DIM * 2);
    __bf16* at_bf  = (__bf16*)carve(M * DIM * 2);

    const int threads = 256;
    pack_s_kernel<<<(BB * NK * DIM + threads - 1) / threads, threads, 0, stream>>>(
        s_x, pos, s_bf);
    cast_f32_to_bf16<<<((int)(M * DIM) + threads - 1) / threads, threads, 0, stream>>>(
        t_x, tx_bf, (int)(M * DIM));
    cast_f32_to_bf16<<<(DIM * DIM + threads - 1) / threads, threads, 0, stream>>>(
        q_w, qw_bf, DIM * DIM);
    cast_f32_to_bf16<<<(2 * DIM * DIM + threads - 1) / threads, threads, 0, stream>>>(
        kv_w, kvw_bf, 2 * DIM * DIM);
    cast_f32_to_bf16<<<(DIM * DIM + threads - 1) / threads, threads, 0, stream>>>(
        proj_w, pw_bf, DIM * DIM);

    // q = (t_x @ q_w^T + q_b) * hd^-0.5   (scale folded)
    gemm_bf16_wmma<__bf16><<<dim3((int)(M / 64), DIM / 256), threads, 0, stream>>>(
        tx_bf, qw_bf, q_b, q_bf, (int)M, DIM, DIM, 0.125f);
    // kv = s @ kv_w^T + kv_b
    gemm_bf16_wmma<__bf16><<<dim3((int)(M / 64), (2 * DIM) / 256), threads, 0, stream>>>(
        s_bf, kvw_bf, kv_b, kv_bf, (int)M, 2 * DIM, DIM, 1.0f);
    // attention (exact grid: 8*12*99 waves = 1188 blocks * 8 waves)
    attn_wmma<<<(BB * HEADS * QTILES) / 8, threads, 0, stream>>>(q_bf, kv_bf, at_bf);
    // out = attn_out @ proj_w^T + proj_b  (fp32 out)
    gemm_bf16_wmma<float><<<dim3((int)(M / 64), DIM / 256), threads, 0, stream>>>(
        at_bf, pw_bf, proj_b, out, (int)M, DIM, DIM, 1.0f);
}
